// Modified_GCN_30408368455787
// MI455X (gfx1250) — matrix operand order repaired and checked by reading, not compiled, not run on hardware
//
#include <hip/hip_runtime.h>
#include <hip/hip_bf16.h>

typedef __attribute__((ext_vector_type(16))) __bf16 v16bf;
typedef __attribute__((ext_vector_type(8)))  float  v8f;

// ---------------------------------------------------------------------------
// Fill a float buffer with a constant (used to init deg=1 for self loops and
// to zero the aggregation buffers each call — harness poisons ws with 0xAA).
// ---------------------------------------------------------------------------
__global__ void fill_kernel(float* __restrict__ p, float v, size_t n) {
    size_t i = (size_t)blockIdx.x * blockDim.x + threadIdx.x;
    if (i < n) p[i] = v;
}

// ---------------------------------------------------------------------------
// Degree: each original edge contributes to both endpoints (graph is made
// undirected); self loop already folded in via deg init = 1.0.
// ---------------------------------------------------------------------------
__global__ void deg_kernel(const int* __restrict__ srcl, const int* __restrict__ dstl,
                           float* __restrict__ deg, int E) {
    int e = blockIdx.x * blockDim.x + threadIdx.x;
    if (e < E) {
        unsafeAtomicAdd(&deg[srcl[e]], 1.0f);
        unsafeAtomicAdd(&deg[dstl[e]], 1.0f);
    }
}

__global__ void dinv_kernel(const float* __restrict__ deg, float* __restrict__ dinv, int n) {
    int i = blockIdx.x * blockDim.x + threadIdx.x;
    if (i < n) {
        float d = deg[i];
        dinv[i] = (d > 0.0f) ? rsqrtf(d) : 0.0f;
    }
}

// ---------------------------------------------------------------------------
// Dense transform H[N][OUTD] = X[N][128] @ W[OUTD][128]^T using bf16 WMMA.
// 256 threads = 8 waves; each wave owns a 16-row M tile, loops over N tiles.
// W^T is staged in LDS as bf16 with a 2-element pad (odd dword stride ->
// conflict-free row reads).
// ---------------------------------------------------------------------------
template <int OUTD>
__global__ void gemm_wmma_bf16(const float* __restrict__ X, const float* __restrict__ W,
                               float* __restrict__ H, int nrows) {
    constexpr int K   = 128;
    constexpr int LDW = OUTD + 2;                 // bf16 elements per LDS row
    __shared__ __bf16 sWt[K * LDW];               // Wt[k][n] = W[n][k]

    const int t = threadIdx.x;                    // 256 threads
    for (int idx = t; idx < OUTD * K; idx += 256) {
        int n = idx / K, k = idx % K;             // coalesced read of W[OUTD][K]
        sWt[k * LDW + n] = (__bf16)W[idx];
    }
    __syncthreads();

    const int wave = t >> 5;                      // 0..7
    const int lane = t & 31;
    const int lm   = lane & 15;
    const int hh   = lane >> 4;                   // lane half (0/1)
    const int m0   = blockIdx.x * 128 + wave * 16;

    int mrow = m0 + lm;
    if (mrow >= nrows) mrow = nrows - 1;          // clamp: keep EXEC all-1s for WMMA
    const float* xr = X + (size_t)mrow * K;

    // Hoist A fragments for all 4 K-tiles (documented 16-bit A layout).
    v16bf a[4];
#pragma unroll
    for (int kb = 0; kb < 4; ++kb) {
        const int base = kb * 32 + hh * 8;
#pragma unroll
        for (int j = 0; j < 8; ++j) {
            a[kb][j]     = (__bf16)xr[base + j];        // K = base + j
            a[kb][j + 8] = (__bf16)xr[base + 16 + j];   // K = base + 16 + j
        }
    }

#pragma unroll
    for (int n0 = 0; n0 < OUTD; n0 += 16) {
        v8f acc = {};
#pragma unroll
        for (int kb = 0; kb < 4; ++kb) {
            // B fragment: lane holds column n0+lm, elements are K in table order (as A).
            union { v16bf v; unsigned u[8]; } bb;
            const __bf16* wcol = sWt + (size_t)(kb * 32 + hh * 8) * LDW + n0 + lm;
#pragma unroll
            for (int j = 0; j < 8; ++j) {
                bb.v[j]     = wcol[(size_t)j * LDW];
                bb.v[j + 8] = wcol[(size_t)(16 + j) * LDW];
            }
            acc = __builtin_amdgcn_wmma_f32_16x16x32_bf16(
                false, a[kb], false, bb.v, (short)0, acc, false, false);
        }
        // D layout: VGPR r -> row m0 + r + 8*hh, col n0 + lm
#pragma unroll
        for (int r = 0; r < 8; ++r) {
            int row = m0 + r + 8 * hh;
            if (row < nrows) H[(size_t)row * OUTD + n0 + lm] = acc[r];
        }
    }
}

// ---------------------------------------------------------------------------
// Edge aggregation: one wave per edge, both directions.
// agg[v] += H[u]*dinv[u]*dinv[v];  agg[u] += H[v]*dinv[u]*dinv[v]
// VEC = F/32 floats per lane (4 for F=128, 2 for F=64). L2-resident atomics.
// ---------------------------------------------------------------------------
template <int F, int VEC>
__global__ void scatter_kernel(const float* __restrict__ H,
                               const int* __restrict__ srcl, const int* __restrict__ dstl,
                               const float* __restrict__ dinv,
                               float* __restrict__ agg, int E) {
    int tid  = blockIdx.x * blockDim.x + threadIdx.x;
    int e    = tid >> 5;
    int lane = tid & 31;
    if (e >= E) return;

    const int u = srcl[e];
    const int v = dstl[e];
    const float w = dinv[u] * dinv[v];
    const int f = lane * VEC;

    const float* hu = H + (size_t)u * F + f;
    const float* hv = H + (size_t)v * F + f;
    float* av = agg + (size_t)v * F + f;
    float* au = agg + (size_t)u * F + f;
#pragma unroll
    for (int i = 0; i < VEC; ++i) {
        unsafeAtomicAdd(&av[i], hu[i] * w);
        unsafeAtomicAdd(&au[i], hv[i] * w);
    }
}

// ---------------------------------------------------------------------------
// Epilogue: add self-loop (H[i]*dinv[i]^2), bias, optional ReLU. In-place on agg.
// ---------------------------------------------------------------------------
template <int F, bool RELU>
__global__ void epilogue_kernel(float* __restrict__ agg, const float* __restrict__ H,
                                const float* __restrict__ dinv, const float* __restrict__ b,
                                int n) {
    size_t tid = (size_t)blockIdx.x * blockDim.x + threadIdx.x;
    if (tid >= (size_t)n * F) return;
    int i = (int)(tid / F);
    int f = (int)(tid % F);
    float s = dinv[i]; s = s * s;
    float val = agg[tid] + H[tid] * s + b[f];
    if (RELU) val = fmaxf(val, 0.0f);
    agg[tid] = val;
}

// ---------------------------------------------------------------------------
// Row-wise log-softmax over 64 classes; one wave32 per node, 2 values/lane,
// shuffle-based max/sum reductions.
// ---------------------------------------------------------------------------
__global__ void logsoftmax64_kernel(const float* __restrict__ Z, float* __restrict__ out, int n) {
    int tid  = blockIdx.x * blockDim.x + threadIdx.x;
    int node = tid >> 5;
    int lane = tid & 31;
    if (node >= n) return;
    const float* z = Z + (size_t)node * 64;
    float a = z[lane];
    float b = z[lane + 32];
    float m = fmaxf(a, b);
#pragma unroll
    for (int o = 16; o > 0; o >>= 1) m = fmaxf(m, __shfl_xor(m, o, 32));
    float s = __expf(a - m) + __expf(b - m);
#pragma unroll
    for (int o = 16; o > 0; o >>= 1) s += __shfl_xor(s, o, 32);
    float l = m + __logf(s);
    out[(size_t)node * 64 + lane]      = a - l;
    out[(size_t)node * 64 + lane + 32] = b - l;
}

// ---------------------------------------------------------------------------
extern "C" void kernel_launch(void* const* d_in, const int* in_sizes, int n_in,
                              void* d_out, int out_size, void* d_ws, size_t ws_size,
                              hipStream_t stream) {
    const float* x  = (const float*)d_in[0];
    const int*   eg = (const int*)  d_in[1];   // [2, E]: out_nodes then in_nodes
    const float* W0 = (const float*)d_in[2];
    const float* b0 = (const float*)d_in[3];
    const float* W1 = (const float*)d_in[4];
    const float* b1 = (const float*)d_in[5];
    const float* W2 = (const float*)d_in[6];
    const float* b2 = (const float*)d_in[7];
    float* out = (float*)d_out;

    const int N = in_sizes[0] / 128;           // 100000 nodes
    const int E = in_sizes[1] / 2;             // 1600000 edges
    const int* srcl = eg;
    const int* dstl = eg + E;

    float* ws      = (float*)d_ws;
    const size_t NF = (size_t)N * 128;
    float* deg   = ws;                         // N
    float* dinv  = ws + N;                     // N
    float* Hbuf  = ws + 2 * (size_t)N;         // N*128
    float* abuf  = Hbuf + NF;                  // N*128  (agg / activation ping)
    float* bbuf  = abuf + NF;                  // N*128  (agg / activation pong)

    const int TB = 256;
    const int gemmBlocks = (N + 127) / 128;
    const int edgeBlocks = (E * 32 + TB - 1) / TB;   // wave per edge
    const int nBlocks    = (N + TB - 1) / TB;

    // --- normalization -----------------------------------------------------
    fill_kernel<<<nBlocks, TB, 0, stream>>>(deg, 1.0f, (size_t)N);   // self loop
    deg_kernel<<<(E + TB - 1) / TB, TB, 0, stream>>>(srcl, dstl, deg, E);
    dinv_kernel<<<nBlocks, TB, 0, stream>>>(deg, dinv, N);

    // --- layer 0: x -> abuf (128) ------------------------------------------
    gemm_wmma_bf16<128><<<gemmBlocks, TB, 0, stream>>>(x, W0, Hbuf, N);
    fill_kernel<<<(unsigned)((NF + TB - 1) / TB), TB, 0, stream>>>(abuf, 0.0f, NF);
    scatter_kernel<128, 4><<<edgeBlocks, TB, 0, stream>>>(Hbuf, srcl, dstl, dinv, abuf, E);
    epilogue_kernel<128, true><<<(unsigned)((NF + TB - 1) / TB), TB, 0, stream>>>(
        abuf, Hbuf, dinv, b0, N);

    // --- layer 1: abuf -> bbuf (128) ---------------------------------------
    gemm_wmma_bf16<128><<<gemmBlocks, TB, 0, stream>>>(abuf, W1, Hbuf, N);
    fill_kernel<<<(unsigned)((NF + TB - 1) / TB), TB, 0, stream>>>(bbuf, 0.0f, NF);
    scatter_kernel<128, 4><<<edgeBlocks, TB, 0, stream>>>(Hbuf, srcl, dstl, dinv, bbuf, E);
    epilogue_kernel<128, true><<<(unsigned)((NF + TB - 1) / TB), TB, 0, stream>>>(
        bbuf, Hbuf, dinv, b1, N);

    // --- layer 2: bbuf -> abuf (64) ----------------------------------------
    const size_t NF2 = (size_t)N * 64;
    gemm_wmma_bf16<64><<<gemmBlocks, TB, 0, stream>>>(bbuf, W2, Hbuf, N);
    fill_kernel<<<(unsigned)((NF2 + TB - 1) / TB), TB, 0, stream>>>(abuf, 0.0f, NF2);
    scatter_kernel<64, 2><<<edgeBlocks, TB, 0, stream>>>(Hbuf, srcl, dstl, dinv, abuf, E);
    epilogue_kernel<64, false><<<(unsigned)((NF2 + TB - 1) / TB), TB, 0, stream>>>(
        abuf, Hbuf, dinv, b2, N);

    // --- log-softmax -------------------------------------------------------
    logsoftmax64_kernel<<<(N * 32 + TB - 1) / TB, TB, 0, stream>>>(abuf, out, N);
}
